// CausalSelfAttention_35141422416488
// MI455X (gfx1250) — compile-verified, matrix operand order checked by reading, not run on hardware
//
#include <hip/hip_runtime.h>
#include <hip/hip_bf16.h>

// ---------------------------------------------------------------------------
// CDNA5 (gfx1250) causal self-attention, bf16 WMMA + flash-attention streaming
// B=4, T=2048, C=1024, H=16, D=64
// ---------------------------------------------------------------------------

typedef __bf16 bf16;
typedef __attribute__((ext_vector_type(4)))  __bf16 bf16x4;
typedef __attribute__((ext_vector_type(16))) __bf16 bf16x16;
typedef __attribute__((ext_vector_type(8)))  float  f32x8;

#define HEADS 16
#define SEQ   2048
#define HDIM  64
#define CH    1024
#define BATCH 4

__device__ inline f32x8 wmma_bf16(bf16x16 a, bf16x16 b, f32x8 c) {
  // D(16x16,f32) = A(16x32,bf16) * B(32x16,bf16) + C
  return __builtin_amdgcn_wmma_f32_16x16x32_bf16(
      /*neg_a=*/false, a, /*neg_b=*/false, b,
      /*c_mod=*/(short)0, c, /*reuse_a=*/false, /*reuse_b=*/false);
}

// Load a 16x32 bf16 fragment (A layout; B layout symmetric with lane = N).
// Per ISA 7.12.2: lane&15 = row; lanes 0-15 hold K = {0..7, 16..23},
// lanes 16-31 hold K = {8..15, 24..31}; two contiguous 16B loads per lane.
__device__ inline bf16x16 load_frag16(const bf16* __restrict__ p, int ld, int lane) {
  int r  = lane & 15;
  int ko = (lane & 16) ? 8 : 0;
  const bf16* row = p + (long)r * ld;
  bf16x16 f;
#pragma unroll
  for (int e = 0; e < 8; ++e) f[e] = row[ko + e];
#pragma unroll
  for (int e = 0; e < 8; ++e) f[8 + e] = row[16 + ko + e];
  return f;
}

// ---------------------------------------------------------------------------
// Kernel 1a: fp32 -> bf16 straight cast (x), vectorized 16B->8B
// ---------------------------------------------------------------------------
__global__ __launch_bounds__(256) void cast_f32_bf16(const float4* __restrict__ src,
                                                     bf16x4* __restrict__ dst, int n4) {
  int i = blockIdx.x * blockDim.x + threadIdx.x;
  int stride = gridDim.x * blockDim.x;
  for (; i < n4; i += stride) {
    float4 v = src[i];
    bf16x4 o;
    o[0] = (bf16)v.x; o[1] = (bf16)v.y; o[2] = (bf16)v.z; o[3] = (bf16)v.w;
    dst[i] = o;
  }
}

// ---------------------------------------------------------------------------
// Kernel 1b: fp32 (K,N) -> bf16 (N,K) LDS-tiled transpose-cast (coalesced both
// sides; 33-stride padding avoids LDS bank conflicts). Block = 32x8.
// ---------------------------------------------------------------------------
__global__ __launch_bounds__(256) void transpose_cast(const float* __restrict__ src,
                                                      bf16* __restrict__ dst,
                                                      int K, int N) {
  __shared__ float tile[32][33];
  int n0 = blockIdx.x * 32, k0 = blockIdx.y * 32;
  int tx = threadIdx.x, ty = threadIdx.y;
#pragma unroll
  for (int i = 0; i < 32; i += 8)
    tile[ty + i][tx] = src[(long)(k0 + ty + i) * N + n0 + tx];
  __syncthreads();
#pragma unroll
  for (int i = 0; i < 32; i += 8)
    dst[(long)(n0 + ty + i) * K + k0 + tx] = (bf16)tile[tx][ty + i];
}

// ---------------------------------------------------------------------------
// Kernel 2: QKV = x @ Wqkv + b.  32x64 output tile per wave (8 accumulators):
// 2 A-frags + 4 B-frags per 8 WMMAs.  Scatters q/k (B,H,T,D) bf16,
// v transposed (B,H,D,T) bf16, k/v fp32 to d_out.  N-tile (64-wide, aligned)
// never crosses a head or a q/k/v section.
// ---------------------------------------------------------------------------
__global__ __launch_bounds__(256) void qkv_gemm(
    const bf16* __restrict__ Xb, const bf16* __restrict__ Wt,
    const float* __restrict__ bias,
    bf16* __restrict__ q_bf, bf16* __restrict__ k_bf, bf16* __restrict__ vt_bf,
    float* __restrict__ k_out, float* __restrict__ v_out) {
  int lane = threadIdx.x & 31;
  int wave = threadIdx.x >> 5;
  int n0 = blockIdx.x * 64;               // column tile in [0, 3C)
  int m0 = blockIdx.y * 256 + wave * 32;  // row tile in [0, B*T)
  f32x8 acc[2][4] = {};
  const bf16* a_base = Xb + (long)m0 * CH;
  const bf16* b_base = Wt + (long)n0 * CH;
  for (int k0 = 0; k0 < CH; k0 += 32) {
    bf16x16 a0 = load_frag16(a_base + k0, CH, lane);
    bf16x16 a1 = load_frag16(a_base + 16 * CH + k0, CH, lane);
    bf16x16 b0 = load_frag16(b_base + 0 * 16 * CH + k0, CH, lane);
    bf16x16 b1 = load_frag16(b_base + 1 * 16 * CH + k0, CH, lane);
    bf16x16 b2 = load_frag16(b_base + 2 * 16 * CH + k0, CH, lane);
    bf16x16 b3 = load_frag16(b_base + 3 * 16 * CH + k0, CH, lane);
    acc[0][0] = wmma_bf16(a0, b0, acc[0][0]);
    acc[0][1] = wmma_bf16(a0, b1, acc[0][1]);
    acc[0][2] = wmma_bf16(a0, b2, acc[0][2]);
    acc[0][3] = wmma_bf16(a0, b3, acc[0][3]);
    acc[1][0] = wmma_bf16(a1, b0, acc[1][0]);
    acc[1][1] = wmma_bf16(a1, b1, acc[1][1]);
    acc[1][2] = wmma_bf16(a1, b2, acc[1][2]);
    acc[1][3] = wmma_bf16(a1, b3, acc[1][3]);
  }
  int half = (lane & 16) ? 8 : 0;
  int sect = n0 >> 10;                    // 0=q 1=k 2=v (uniform per wave)
  int h = (n0 & (CH - 1)) >> 6;           // head (uniform per wave)
  float bv[4];
#pragma unroll
  for (int j = 0; j < 4; ++j) bv[j] = bias[n0 + j * 16 + (lane & 15)];
#pragma unroll
  for (int i = 0; i < 2; ++i) {
#pragma unroll
    for (int r = 0; r < 8; ++r) {
      int m = m0 + i * 16 + r + half;
      int bb = m >> 11;                   // batch
      int t  = m & (SEQ - 1);
      long bht = (long)(bb * HEADS + h) * SEQ + t;
#pragma unroll
      for (int j = 0; j < 4; ++j) {
        int d = j * 16 + (lane & 15);
        float val = acc[i][j][r] + bv[j];
        if (sect == 0) {
          q_bf[bht * HDIM + d] = (bf16)val;
        } else if (sect == 1) {
          k_bf[bht * HDIM + d] = (bf16)val;
          k_out[bht * HDIM + d] = val;
        } else {
          vt_bf[((long)(bb * HEADS + h) * HDIM + d) * SEQ + t] = (bf16)val;
          v_out[bht * HDIM + d] = val;
        }
      }
    }
  }
}

// ---------------------------------------------------------------------------
// Kernel 3: flash attention. Each wave owns a 16-row query tile of one (b,h),
// streams 32-key chunks with online softmax; P transposed via per-wave LDS.
// K/V bf16 (32 MB total) stay L2-resident across the O(T^2) inner loop.
// Row-sum kept as per-lane partials (alpha is lane-uniform per row) and
// reduced once at the end -> saves 32 shuffles per key chunk.
// ---------------------------------------------------------------------------
__global__ __launch_bounds__(256) void flash_attn(
    const bf16* __restrict__ q_all, const bf16* __restrict__ k_all,
    const bf16* __restrict__ vt_all, bf16* __restrict__ attn_bf) {
  __shared__ bf16 plds[8][16 * 32];   // 1 KB per wave
  int lane = threadIdx.x & 31;
  int wave = threadIdx.x >> 5;
  int bh = blockIdx.x;                 // 0..63
  int qt = blockIdx.y * 8 + wave;      // 0..127 query tile
  int t0 = qt * 16;
  int b = bh >> 4, h = bh & 15;

  const bf16* qb = q_all + (long)bh * SEQ * HDIM;
  const bf16* kb = k_all + (long)bh * SEQ * HDIM;
  const bf16* vb = vt_all + (long)bh * HDIM * SEQ;

  bf16x16 qa0 = load_frag16(qb + (long)t0 * HDIM + 0,  HDIM, lane);
  bf16x16 qa1 = load_frag16(qb + (long)t0 * HDIM + 32, HDIM, lane);

  int half = (lane & 16) ? 8 : 0;
  float mrow[8], lrow[8];              // lrow: per-lane partial sums
#pragma unroll
  for (int r = 0; r < 8; ++r) { mrow[r] = -1e30f; lrow[r] = 0.0f; }
  f32x8 o0 = {}, o1 = {}, o2 = {}, o3 = {};
  const float scale = 0.125f;          // 1/sqrt(64)

  for (int k0 = 0; k0 <= t0 + 15; k0 += 32) {
    // S = Q (16xD) @ K^T (Dx32): two 16x16 logit tiles
    bf16x16 kb00 = load_frag16(kb + (long)k0 * HDIM + 0,         HDIM, lane);
    bf16x16 kb01 = load_frag16(kb + (long)k0 * HDIM + 32,        HDIM, lane);
    bf16x16 kb10 = load_frag16(kb + (long)(k0 + 16) * HDIM + 0,  HDIM, lane);
    bf16x16 kb11 = load_frag16(kb + (long)(k0 + 16) * HDIM + 32, HDIM, lane);
    f32x8 s0 = {}, s1 = {};
    s0 = wmma_bf16(qa0, kb00, s0);
    s0 = wmma_bf16(qa1, kb01, s0);
    s1 = wmma_bf16(qa0, kb10, s1);
    s1 = wmma_bf16(qa1, kb11, s1);

    int key0 = k0 + (lane & 15);
    int key1 = key0 + 16;
#pragma unroll
    for (int r = 0; r < 8; ++r) {
      int row = t0 + r + half;
      float v0 = (key0 <= row) ? s0[r] * scale : -1e30f;
      float v1 = (key1 <= row) ? s1[r] * scale : -1e30f;
      // row max: 16-lane xor reduction stays inside each wave half,
      // matching the C/D fragment layout (lane half selects M+8).
      float tmax = fmaxf(v0, v1);
      tmax = fmaxf(tmax, __shfl_xor(tmax, 1, 32));
      tmax = fmaxf(tmax, __shfl_xor(tmax, 2, 32));
      tmax = fmaxf(tmax, __shfl_xor(tmax, 4, 32));
      tmax = fmaxf(tmax, __shfl_xor(tmax, 8, 32));
      float mnew = fmaxf(mrow[r], tmax);
      float alpha = __expf(mrow[r] - mnew);   // lane-uniform per row
      float p0 = __expf(v0 - mnew);
      float p1 = __expf(v1 - mnew);
      lrow[r] = lrow[r] * alpha + (p0 + p1);  // deferred reduction
      mrow[r] = mnew;
      o0[r] *= alpha; o1[r] *= alpha; o2[r] *= alpha; o3[r] *= alpha;
      // stage P (16x32 bf16) for A-fragment reload
      plds[wave][(r + half) * 32 + (lane & 15)]      = (bf16)p0;
      plds[wave][(r + half) * 32 + 16 + (lane & 15)] = (bf16)p1;
    }
    asm volatile("s_wait_dscnt 0" ::: "memory");  // same-wave DS ordering
    bf16x16 pa  = load_frag16(&plds[wave][0], 32, lane);
    // O += P (16x32) @ V (32xD): V stored transposed (D,T) -> contiguous keys
    bf16x16 vb0 = load_frag16(vb + (long)(0 * 16) * SEQ + k0, SEQ, lane);
    bf16x16 vb1 = load_frag16(vb + (long)(1 * 16) * SEQ + k0, SEQ, lane);
    bf16x16 vb2 = load_frag16(vb + (long)(2 * 16) * SEQ + k0, SEQ, lane);
    bf16x16 vb3 = load_frag16(vb + (long)(3 * 16) * SEQ + k0, SEQ, lane);
    o0 = wmma_bf16(pa, vb0, o0);
    o1 = wmma_bf16(pa, vb1, o1);
    o2 = wmma_bf16(pa, vb2, o2);
    o3 = wmma_bf16(pa, vb3, o3);
    asm volatile("" ::: "memory");  // don't let next-iter LDS stores hoist
  }

#pragma unroll
  for (int r = 0; r < 8; ++r) {
    // single final row-sum reduction
    float s = lrow[r];
    s += __shfl_xor(s, 1, 32);
    s += __shfl_xor(s, 2, 32);
    s += __shfl_xor(s, 4, 32);
    s += __shfl_xor(s, 8, 32);
    float inv = 1.0f / s;
    int t = t0 + r + half;
    long base = ((long)(b * SEQ + t)) * CH + h * HDIM + (lane & 15);
    attn_bf[base + 0]  = (bf16)(o0[r] * inv);
    attn_bf[base + 16] = (bf16)(o1[r] * inv);
    attn_bf[base + 32] = (bf16)(o2[r] * inv);
    attn_bf[base + 48] = (bf16)(o3[r] * inv);
  }
}

// ---------------------------------------------------------------------------
// Kernel 4: y = attn_out @ Wout + bout (fp32 out). 32x64 tile per wave.
// ---------------------------------------------------------------------------
__global__ __launch_bounds__(256) void out_gemm(
    const bf16* __restrict__ Ab, const bf16* __restrict__ Wt,
    const float* __restrict__ bias, float* __restrict__ y) {
  int lane = threadIdx.x & 31;
  int wave = threadIdx.x >> 5;
  int n0 = blockIdx.x * 64;
  int m0 = blockIdx.y * 256 + wave * 32;
  f32x8 acc[2][4] = {};
  const bf16* a_base = Ab + (long)m0 * CH;
  const bf16* b_base = Wt + (long)n0 * CH;
  for (int k0 = 0; k0 < CH; k0 += 32) {
    bf16x16 a0 = load_frag16(a_base + k0, CH, lane);
    bf16x16 a1 = load_frag16(a_base + 16 * CH + k0, CH, lane);
    bf16x16 b0 = load_frag16(b_base + 0 * 16 * CH + k0, CH, lane);
    bf16x16 b1 = load_frag16(b_base + 1 * 16 * CH + k0, CH, lane);
    bf16x16 b2 = load_frag16(b_base + 2 * 16 * CH + k0, CH, lane);
    bf16x16 b3 = load_frag16(b_base + 3 * 16 * CH + k0, CH, lane);
    acc[0][0] = wmma_bf16(a0, b0, acc[0][0]);
    acc[0][1] = wmma_bf16(a0, b1, acc[0][1]);
    acc[0][2] = wmma_bf16(a0, b2, acc[0][2]);
    acc[0][3] = wmma_bf16(a0, b3, acc[0][3]);
    acc[1][0] = wmma_bf16(a1, b0, acc[1][0]);
    acc[1][1] = wmma_bf16(a1, b1, acc[1][1]);
    acc[1][2] = wmma_bf16(a1, b2, acc[1][2]);
    acc[1][3] = wmma_bf16(a1, b3, acc[1][3]);
  }
  int half = (lane & 16) ? 8 : 0;
  float bv[4];
#pragma unroll
  for (int j = 0; j < 4; ++j) bv[j] = bias[n0 + j * 16 + (lane & 15)];
#pragma unroll
  for (int i = 0; i < 2; ++i) {
#pragma unroll
    for (int r = 0; r < 8; ++r) {
      int m = m0 + i * 16 + r + half;
#pragma unroll
      for (int j = 0; j < 4; ++j)
        y[(long)m * CH + n0 + j * 16 + (lane & 15)] = acc[i][j][r] + bv[j];
    }
  }
}

// ---------------------------------------------------------------------------
// Launcher
// ---------------------------------------------------------------------------
extern "C" void kernel_launch(void* const* d_in, const int* in_sizes, int n_in,
                              void* d_out, int out_size, void* d_ws, size_t ws_size,
                              hipStream_t stream) {
  const float* x    = (const float*)d_in[0];   // (B,T,C)
  const float* Wqkv = (const float*)d_in[1];   // (C,3C)
  const float* bqkv = (const float*)d_in[2];   // (3C)
  const float* Wout = (const float*)d_in[3];   // (C,C)
  const float* bout = (const float*)d_in[4];   // (C)

  const long NX  = (long)BATCH * SEQ * CH;     // 8388608 (also = B*H*T*D)
  float* y     = (float*)d_out;                // (B,T,C)
  float* k_out = y + NX;                       // (B,H,T,D)
  float* v_out = y + 2 * NX;                   // (B,H,T,D)

  char* ws = (char*)d_ws;
  bf16* x_bf    = (bf16*)(ws);                         // 16 MB
  bf16* wqkv_t  = (bf16*)(ws + 16777216);              // 6 MB  (3C,C)
  bf16* wout_t  = (bf16*)(ws + 23068672);              // 2 MB  (C,C)
  bf16* q_bf    = (bf16*)(ws + 25165824);              // 16 MB (B,H,T,D)
  bf16* k_bf    = (bf16*)(ws + 41943040);              // 16 MB (B,H,T,D)
  bf16* vt_bf   = (bf16*)(ws + 58720256);              // 16 MB (B,H,D,T)
  bf16* attn_bf = (bf16*)(ws + 75497472);              // 16 MB (B,T,C)

  cast_f32_bf16<<<2048, 256, 0, stream>>>((const float4*)x, (bf16x4*)x_bf,
                                          (int)(NX >> 2));
  transpose_cast<<<dim3(96, 32), dim3(32, 8), 0, stream>>>(Wqkv, wqkv_t, CH, 3 * CH);
  transpose_cast<<<dim3(32, 32), dim3(32, 8), 0, stream>>>(Wout, wout_t, CH, CH);

  // M = B*T = 8192 rows (256/block), N = 3C = 3072 cols (64/wave)
  qkv_gemm<<<dim3(48, 32), 256, 0, stream>>>(x_bf, wqkv_t, bqkv,
                                             q_bf, k_bf, vt_bf, k_out, v_out);

  // 64 (b,h) pairs x 128 query tiles (8 tiles/block)
  flash_attn<<<dim3(64, 16), 256, 0, stream>>>(q_bf, k_bf, vt_bf, attn_bf);

  // M = 8192, N = 1024
  out_gemm<<<dim3(16, 32), 256, 0, stream>>>(attn_bf, wout_t, bout, y);
}
